// SparseSelfAttention_79156247265914
// MI455X (gfx1250) — matrix operand order repaired and checked by reading, not compiled, not run on hardware
//
#include <hip/hip_runtime.h>
#include <hip/hip_bf16.h>
#include <math.h>

// ---------------------------------------------------------------------------
// MoE sparse self-attention for MI455X (gfx1250, wave32, WMMA f32_16x16x32_f16)
//
// Only the top-2 experts per batch contribute to the output (dispatch mask is
// exactly {0,1} with coefficient 1.0), so all heavy GEMMs are launched per
// (batch, slot) pair with the expert id read from a device-side table.
// ---------------------------------------------------------------------------

typedef _Float16 v16h __attribute__((ext_vector_type(16)));
typedef _Float16 v8h  __attribute__((ext_vector_type(8)));
typedef float    v8f  __attribute__((ext_vector_type(8)));

static constexpr int  E_   = 8;
static constexpr int  B_   = 4;
static constexpr int  S_   = 1024;
static constexpr int  D_   = 1024;
static constexpr int  H_   = 1024;
static constexpr int  TOPK = 2;
static constexpr float SCALE_ = 0.03125f;   // 1/sqrt(1024)

static constexpr int BM = 64, BN = 64, BK = 32;
static constexpr int LDT = BK + 8;          // LDS row pitch (halves), 16B aligned, anti-bank-conflict

// out modes
static constexpr int OUT_F32  = 0;  // f32 row-major
static constexpr int OUT_F16  = 1;  // f16 row-major
static constexpr int OUT_F16T = 2;  // f16 transposed ([n][m])

// ---------------------------------------------------------------------------
// Router: logits[b,e] = X[b].flat · switch_W[:,e] + switch_b[e]
// Deterministic 2-stage reduction (no float atomics -> bit-stable replays).
// ---------------------------------------------------------------------------
static constexpr int RCHUNKS = 32;
static constexpr int RROWS   = (S_ * D_) / RCHUNKS;   // 32768 rows per chunk

__global__ __launch_bounds__(256)
void router_partial(const float* __restrict__ X, const float* __restrict__ W,
                    float* __restrict__ partial) {
  const int chunk = blockIdx.x;
  const int b     = blockIdx.y;
  const int t     = threadIdx.x;
  const float* Xb = X + ((size_t)b << 20);
  float acc[8];
#pragma unroll
  for (int e = 0; e < 8; ++e) acc[e] = 0.f;
  const size_t base = (size_t)chunk * RROWS;
  for (int j = 0; j < RROWS / 256; ++j) {
    size_t i = base + (size_t)j * 256 + t;
    float x = Xb[i];
    const float4* w4 = (const float4*)(W + i * 8);
    float4 wa = w4[0], wb = w4[1];
    acc[0] += x * wa.x; acc[1] += x * wa.y; acc[2] += x * wa.z; acc[3] += x * wa.w;
    acc[4] += x * wb.x; acc[5] += x * wb.y; acc[6] += x * wb.z; acc[7] += x * wb.w;
  }
  __shared__ float red[256];
  for (int e = 0; e < 8; ++e) {
    red[t] = acc[e];
    __syncthreads();
    for (int s = 128; s > 0; s >>= 1) {
      if (t < s) red[t] += red[t + s];
      __syncthreads();
    }
    if (t == 0) partial[((size_t)chunk * B_ + b) * 8 + e] = red[0];
    __syncthreads();
  }
}

__global__ void router_topk(const float* __restrict__ partial,
                            const float* __restrict__ switch_b,
                            int* __restrict__ eidx) {
  __shared__ float slog[B_ * E_];
  const int t = threadIdx.x;
  if (t < B_ * E_) {
    const int b = t >> 3, e = t & 7;
    float s = switch_b[e];
    for (int c = 0; c < RCHUNKS; ++c) s += partial[((size_t)c * B_ + b) * 8 + e];
    slog[t] = s;   // softmax is monotone -> top-k on logits is identical
  }
  __syncthreads();
  if (t < B_) {
    const float* l = &slog[t * E_];
    int e1 = 0;
    for (int e = 1; e < E_; ++e) if (l[e] > l[e1]) e1 = e;
    int e2 = (e1 == 0) ? 1 : 0;
    for (int e = 0; e < E_; ++e) if (e != e1 && l[e] > l[e2]) e2 = e;
    eidx[t * TOPK + 0] = e1;
    eidx[t * TOPK + 1] = e2;
  }
}

// ---------------------------------------------------------------------------
// fp32 -> fp16 elementwise convert (for X)
// ---------------------------------------------------------------------------
__global__ __launch_bounds__(256)
void convert_f32_f16(const float* __restrict__ src, _Float16* __restrict__ dst) {
  size_t i = ((size_t)blockIdx.x * 256 + threadIdx.x) * 4;
  float4 v = *(const float4*)(src + i);
  dst[i + 0] = (_Float16)v.x;
  dst[i + 1] = (_Float16)v.y;
  dst[i + 2] = (_Float16)v.z;
  dst[i + 3] = (_Float16)v.w;
}

// ---------------------------------------------------------------------------
// WMMA GEMM core helpers: one k-step of tile staging + 2x2 WMMA per wave.
// A operand: fp16 row-major [M][K].
// B flavor 0: fp16 global stored [N][K]  (already "transposed": K rows / V^T)
// B flavor 1: fp32 global stored [K][N]  (weights) -> convert+transpose to LDS
// LDS layouts: Al[m][k], Bl[n][k] with pitch LDT.
// ---------------------------------------------------------------------------
__device__ __forceinline__ v16h make_frag(v8h lo, v8h hi) {
  return __builtin_shufflevector(lo, hi, 0, 1, 2, 3, 4, 5, 6, 7,
                                 8, 9, 10, 11, 12, 13, 14, 15);
}

template <int BF>
__device__ __forceinline__ void gemm_ksteps(const _Float16* __restrict__ Ag,
                                            const _Float16* __restrict__ Bg16,
                                            const float* __restrict__ Bg32,
                                            _Float16* Al, _Float16* Bl,
                                            int m0, int n0, v8f acc[2][2]) {
  const int t     = threadIdx.x;
  const int wave  = t >> 5, lane = t & 31;
  const int wm    = (wave >> 1) * 32, wn = (wave & 1) * 32;
  const int lhalf = lane >> 4, l16 = lane & 15;

  for (int kt = 0; kt < 1024 / BK; ++kt) {
    const int k0 = kt * BK;
    // ---- stage A tile (64 x 32 halves): 128 threads x 32B
    {
      const int r = t >> 1, c = (t & 1) * 16;
      const uint4* src = (const uint4*)(Ag + (size_t)(m0 + r) * 1024 + k0 + c);
      uint4* dst = (uint4*)(&Al[r * LDT + c]);
      dst[0] = src[0];
      dst[1] = src[1];
      if (k0 + BK < 1024)
        __builtin_prefetch((const void*)(Ag + (size_t)(m0 + r) * 1024 + k0 + BK + c), 0, 3);
    }
    // ---- stage B tile
    if constexpr (BF == 0) {
      const int r = t >> 1, c = (t & 1) * 16;
      const uint4* src = (const uint4*)(Bg16 + (size_t)(n0 + r) * 1024 + k0 + c);
      uint4* dst = (uint4*)(&Bl[r * LDT + c]);
      dst[0] = src[0];
      dst[1] = src[1];
      if (k0 + BK < 1024)
        __builtin_prefetch((const void*)(Bg16 + (size_t)(n0 + r) * 1024 + k0 + BK + c), 0, 3);
    } else {
#pragma unroll
      for (int it = 0; it < 4; ++it) {
        const int idx = t * 4 + it;
        const int kk  = idx >> 4;          // 0..31  (k within tile)
        const int nc  = (idx & 15) * 4;    // 0..60  (n within tile)
        const float4 w = *(const float4*)(Bg32 + (size_t)(k0 + kk) * 1024 + n0 + nc);
        const float wv[4] = {w.x, w.y, w.z, w.w};
#pragma unroll
        for (int j = 0; j < 4; ++j)
          Bl[(nc + j) * LDT + kk] = (_Float16)wv[j];
      }
      if (k0 + BK < 1024)
        __builtin_prefetch((const void*)(Bg32 + (size_t)(k0 + BK) * 1024 + n0 + (t & 15) * 4), 0, 3);
    }
    __syncthreads();

    // ---- fragments per ISA VGPR layouts (wave32)
    v16h afr[2], bfr[2];
#pragma unroll
    for (int mt = 0; mt < 2; ++mt) {
      const _Float16* base = &Al[(wm + mt * 16 + l16) * LDT];
      v8h lo = *(const v8h*)(base + lhalf * 8);        // K = 8*lhalf .. +7
      v8h hi = *(const v8h*)(base + 16 + lhalf * 8);   // K = 16+8*lhalf .. +7
      afr[mt] = make_frag(lo, hi);
    }
#pragma unroll
    for (int nt = 0; nt < 2; ++nt) {
      const _Float16* base = &Bl[(wn + nt * 16 + l16) * LDT];
      v8h lo = *(const v8h*)(base + lhalf * 8);
      v8h hi = *(const v8h*)(base + 16 + lhalf * 8);
      bfr[nt] = make_frag(lo, hi);
    }
#pragma unroll
    for (int mt = 0; mt < 2; ++mt)
#pragma unroll
      for (int nt = 0; nt < 2; ++nt)
        acc[mt][nt] = __builtin_amdgcn_wmma_f32_16x16x32_f16(
            false, afr[mt], false, bfr[nt], (short)0, acc[mt][nt], false, false);
    __syncthreads();
  }
}

// ---------------------------------------------------------------------------
// Generic per-(b,slot) WMMA GEMM: C = A(f16) * B, 1024^3, out per outMode.
// aMode: 1 -> A indexed by gz; 0 -> A indexed by batch (gz>>1)
// bMode: 1 -> B indexed by gz; 0 -> B indexed by expert eidx[gz]
// ---------------------------------------------------------------------------
template <int BF>
__global__ __launch_bounds__(128)
void wmma_gemm(const _Float16* __restrict__ A, int aMode,
               const void* __restrict__ Bv, int bMode,
               void* __restrict__ Out, int outMode, float scale,
               const int* __restrict__ eidx) {
  __shared__ __align__(16) _Float16 Al[BM * LDT];
  __shared__ __align__(16) _Float16 Bl[BN * LDT];
  const int gz = blockIdx.z;
  const int m0 = blockIdx.x * BM, n0 = blockIdx.y * BN;

  const size_t ai = aMode ? (size_t)gz : (size_t)(gz >> 1);
  const size_t bi = bMode ? (size_t)gz : (size_t)eidx[gz];
  const _Float16* Ag   = A + (ai << 20);
  const _Float16* Bg16 = (const _Float16*)Bv + (bi << 20);
  const float*    Bg32 = (const float*)Bv + (bi << 20);

  v8f acc[2][2];
#pragma unroll
  for (int mt = 0; mt < 2; ++mt)
#pragma unroll
    for (int nt = 0; nt < 2; ++nt)
#pragma unroll
      for (int i = 0; i < 8; ++i) acc[mt][nt][i] = 0.0f;

  gemm_ksteps<BF>(Ag, Bg16, Bg32, Al, Bl, m0, n0, acc);

  // ---- epilogue: C/D layout -> VGPR r holds row (8*lhalf + r), col l16
  const int lane = threadIdx.x & 31, wave = threadIdx.x >> 5;
  const int wm = (wave >> 1) * 32, wn = (wave & 1) * 32;
  const int lhalf = lane >> 4, l16 = lane & 15;
  const size_t off = (size_t)gz << 20;
#pragma unroll
  for (int mt = 0; mt < 2; ++mt)
#pragma unroll
    for (int nt = 0; nt < 2; ++nt)
#pragma unroll
      for (int r = 0; r < 8; ++r) {
        const int gm = m0 + wm + mt * 16 + lhalf * 8 + r;
        const int gn = n0 + wn + nt * 16 + l16;
        const float v = acc[mt][nt][r] * scale;
        if (outMode == OUT_F32)
          ((float*)Out)[off + (size_t)gm * 1024 + gn] = v;
        else if (outMode == OUT_F16)
          ((_Float16*)Out)[off + (size_t)gm * 1024 + gn] = (_Float16)v;
        else
          ((_Float16*)Out)[off + (size_t)gn * 1024 + gm] = (_Float16)v;
      }
}

// ---------------------------------------------------------------------------
// Output projection: y[b] = sum_{slot} ctx[b,slot] @ Wff[eidx[b,slot]]
// Two K-segments accumulated into the same accumulators -> single f32 store.
// ---------------------------------------------------------------------------
__global__ __launch_bounds__(128)
void wmma_gemm_out2(const _Float16* __restrict__ ctxh,
                    const float* __restrict__ Wff,
                    const int* __restrict__ eidx,
                    float* __restrict__ y) {
  __shared__ __align__(16) _Float16 Al[BM * LDT];
  __shared__ __align__(16) _Float16 Bl[BN * LDT];
  const int b  = blockIdx.z;
  const int m0 = blockIdx.x * BM, n0 = blockIdx.y * BN;

  v8f acc[2][2];
#pragma unroll
  for (int mt = 0; mt < 2; ++mt)
#pragma unroll
    for (int nt = 0; nt < 2; ++nt)
#pragma unroll
      for (int i = 0; i < 8; ++i) acc[mt][nt][i] = 0.0f;

  for (int seg = 0; seg < TOPK; ++seg) {
    const int e = eidx[b * TOPK + seg];
    const _Float16* Ag = ctxh + ((size_t)(b * TOPK + seg) << 20);
    const float*    Bg = Wff + ((size_t)e << 20);
    gemm_ksteps<1>(Ag, nullptr, Bg, Al, Bl, m0, n0, acc);
  }

  const int lane = threadIdx.x & 31, wave = threadIdx.x >> 5;
  const int wm = (wave >> 1) * 32, wn = (wave & 1) * 32;
  const int lhalf = lane >> 4, l16 = lane & 15;
  const size_t off = (size_t)b << 20;
#pragma unroll
  for (int mt = 0; mt < 2; ++mt)
#pragma unroll
    for (int nt = 0; nt < 2; ++nt)
#pragma unroll
      for (int r = 0; r < 8; ++r) {
        const int gm = m0 + wm + mt * 16 + lhalf * 8 + r;
        const int gn = n0 + wn + nt * 16 + l16;
        y[off + (size_t)gm * 1024 + gn] = acc[mt][nt][r];
      }
}

// ---------------------------------------------------------------------------
// Fused per-row: A[b,s,:] = softmax(scores[b,0,s,:]) + softmax(scores[b,1,s,:])
// ---------------------------------------------------------------------------
__device__ __forceinline__ float block_max(float v, float* red) {
  const int t = threadIdx.x;
  red[t] = v; __syncthreads();
  for (int s = 128; s > 0; s >>= 1) { if (t < s) red[t] = fmaxf(red[t], red[t + s]); __syncthreads(); }
  float r = red[0]; __syncthreads(); return r;
}
__device__ __forceinline__ float block_sum(float v, float* red) {
  const int t = threadIdx.x;
  red[t] = v; __syncthreads();
  for (int s = 128; s > 0; s >>= 1) { if (t < s) red[t] += red[t + s]; __syncthreads(); }
  float r = red[0]; __syncthreads(); return r;
}

__global__ __launch_bounds__(256)
void softmax_combine(const float* __restrict__ sc, _Float16* __restrict__ Ah) {
  __shared__ float red[256];
  const int s = blockIdx.x, b = blockIdx.y, t = threadIdx.x;
  const float* r0 = sc + (((size_t)(b * TOPK + 0) * 1024 + s) << 10);
  const float* r1 = sc + (((size_t)(b * TOPK + 1) * 1024 + s) << 10);
  float4 v0 = *(const float4*)(r0 + t * 4);
  float4 v1 = *(const float4*)(r1 + t * 4);

  float M0 = block_max(fmaxf(fmaxf(v0.x, v0.y), fmaxf(v0.z, v0.w)), red);
  float p0[4] = {__expf(v0.x - M0), __expf(v0.y - M0), __expf(v0.z - M0), __expf(v0.w - M0)};
  float Z0 = block_sum(p0[0] + p0[1] + p0[2] + p0[3], red);

  float M1 = block_max(fmaxf(fmaxf(v1.x, v1.y), fmaxf(v1.z, v1.w)), red);
  float p1[4] = {__expf(v1.x - M1), __expf(v1.y - M1), __expf(v1.z - M1), __expf(v1.w - M1)};
  float Z1 = block_sum(p1[0] + p1[1] + p1[2] + p1[3], red);

  const float i0 = 1.0f / Z0, i1 = 1.0f / Z1;
  _Float16* out = Ah + (((size_t)(b * 1024 + s)) << 10) + t * 4;
#pragma unroll
  for (int j = 0; j < 4; ++j) out[j] = (_Float16)(p0[j] * i0 + p1[j] * i1);
}

// ---------------------------------------------------------------------------
// Host-side orchestration (graph-capture safe: static launches, device routing)
// ---------------------------------------------------------------------------
extern "C" void kernel_launch(void* const* d_in, const int* in_sizes, int n_in,
                              void* d_out, int out_size, void* d_ws, size_t ws_size,
                              hipStream_t stream) {
  const float* X   = (const float*)d_in[0];
  // d_in[1] = mask (unused by reference)
  const float* sW  = (const float*)d_in[2];
  const float* sb  = (const float*)d_in[3];
  const float* Wq  = (const float*)d_in[4];
  const float* Wk  = (const float*)d_in[6];
  const float* Wv  = (const float*)d_in[8];
  const float* Wff = (const float*)d_in[10];
  float* y = (float*)d_out;

  // workspace layout (~88 MB peak, with aliasing)
  char* ws = (char*)d_ws;
  int*       eidx    = (int*)ws;                                   // 32 B
  float*     partial = (float*)(ws + 64);                          // 4 KB
  size_t o = 8192;
  _Float16*  Xh      = (_Float16*)(ws + o); o += (size_t)8  << 20; // [B][S][D]
  _Float16*  Vth     = (_Float16*)(ws + o); o += (size_t)16 << 20; // [B*2][H][S]  (transposed)
  _Float16*  Qh      = (_Float16*)(ws + o); o += (size_t)16 << 20; // [B*2][S][H]
  _Float16*  Kh      = (_Float16*)(ws + o); o += (size_t)16 << 20; // [B*2][S][H]
  float*     scoresF = (float*)(ws + o);    o += (size_t)32 << 20; // [B*2][S][S]
  _Float16*  Ah      = Qh;   // Q dead after scores; A needs 8 MB
  _Float16*  ctxh    = Kh;   // K dead after scores; ctx needs 16 MB

  const dim3 gGemm(S_ / BM, H_ / BN, B_ * TOPK);

  // 1) router (deterministic)
  router_partial<<<dim3(RCHUNKS, B_), 256, 0, stream>>>(X, sW, partial);
  router_topk<<<1, 32, 0, stream>>>(partial, sb, eidx);

  // 2) X -> fp16
  convert_f32_f16<<<(B_ * S_ * D_) / 1024, 256, 0, stream>>>(X, Xh);

  // 3) Q/K/V projections for the 8 live (b,slot) pairs only
  wmma_gemm<1><<<gGemm, 128, 0, stream>>>(Xh, 0, (const void*)Wq, 0, (void*)Qh,  OUT_F16,  1.0f, eidx);
  wmma_gemm<1><<<gGemm, 128, 0, stream>>>(Xh, 0, (const void*)Wk, 0, (void*)Kh,  OUT_F16,  1.0f, eidx);
  wmma_gemm<1><<<gGemm, 128, 0, stream>>>(Xh, 0, (const void*)Wv, 0, (void*)Vth, OUT_F16T, 1.0f, eidx);

  // 4) scores = Q K^T * scale   (B operand = K rows, already [N][K])
  wmma_gemm<0><<<gGemm, 128, 0, stream>>>(Qh, 1, (const void*)Kh, 1, (void*)scoresF, OUT_F32, SCALE_, eidx);

  // 5) A = softmax(slot0) + softmax(slot1)   (mask coefficients are exactly 1)
  softmax_combine<<<dim3(S_, B_), 256, 0, stream>>>(scoresF, Ah);

  // 6) ctx = A @ V   (B operand = V^T, [N=h][K=t])
  wmma_gemm<0><<<gGemm, 128, 0, stream>>>(Ah, 0, (const void*)Vth, 1, (void*)ctxh, OUT_F16, 1.0f, eidx);

  // 7) y[b] = sum_slot ctx @ Wff[e]   (single f32 store, no atomics)
  wmma_gemm_out2<<<dim3(S_ / BM, D_ / BN, B_), 128, 0, stream>>>(ctxh, Wff, eidx, y);
}
